// AdditiveAttention_28217935134788
// MI455X (gfx1250) — matrix operand order repaired and checked by reading, not compiled, not run on hardware
//
#include <hip/hip_runtime.h>
#include <hip/hip_bf16.h>

#define B_SZ 32
#define S_SZ 2048
#define D_SZ 512
#define H_SZ 512

typedef __attribute__((ext_vector_type(16))) __bf16       v16bf;
typedef __attribute__((ext_vector_type(8)))  float        v8f;
typedef __attribute__((ext_vector_type(4)))  unsigned int u32x4;
typedef __attribute__((ext_vector_type(4)))  int          i32x4;

union Frag { u32x4 u[2]; v16bf v; };

__device__ __forceinline__ unsigned short f2bf_rne(float f) {
    unsigned u = __float_as_uint(f);
    u += 0x7FFFu + ((u >> 16) & 1u);
    return (unsigned short)(u >> 16);
}
__device__ __forceinline__ float bf2f(unsigned short h) {
    return __uint_as_float(((unsigned)h) << 16);
}

// Branchless tanh: no exec-mask divergence between WMMA bursts.
// tanh(x) = sign(x) * (1 - 2/(e^{2|x|}+1)); exp overflow -> inf -> rcp -> 0 -> 1.
__device__ __forceinline__ float fast_tanh(float x) {
    float ax = fabsf(x);
    float e2 = __expf(ax + ax);                       // v_exp_f32 based, branchless
    float t  = 1.0f - 2.0f * __builtin_amdgcn_rcpf(e2 + 1.0f);
    return copysignf(t, x);
}

// ---- gfx1250 async global->LDS path (guarded; falls back cleanly) ----------
#if defined(__HIP_DEVICE_COMPILE__) && defined(__gfx1250__) && \
    __has_builtin(__builtin_amdgcn_global_load_async_to_lds_b128)
#define USE_ASYNC 1
typedef __attribute__((address_space(1))) i32x4 g_i32x4;   // prints as "int4 __device__*"
typedef __attribute__((address_space(3))) i32x4 l_i32x4;   // LDS
#define ASYNC_CP16(dst, src) \
    __builtin_amdgcn_global_load_async_to_lds_b128((g_i32x4*)(src), (l_i32x4*)(dst), 0, 0)
#if __has_builtin(__builtin_amdgcn_s_wait_asynccnt)
#define WAIT_ASYNC(n) __builtin_amdgcn_s_wait_asynccnt(n)
#else
#define WAIT_ASYNC(n) asm volatile("s_wait_asynccnt %0" :: "i"(n) : "memory")
#endif
#else
#define USE_ASYNC 0
#endif

// ---------------------------------------------------------------------------
// Kernel 1: q_proj[b,h] = sum_d query[b,d] * Wq[h,d]   (32x512, tiny)
// ---------------------------------------------------------------------------
__global__ __launch_bounds__(256) void qproj_kernel(
    const float* __restrict__ query, const float* __restrict__ Wq,
    float* __restrict__ qp)
{
    int idx = blockIdx.x * 256 + threadIdx.x;      // 16384 total
    int b = idx >> 9, h = idx & 511;
    const float4* qr = reinterpret_cast<const float4*>(query + (size_t)b * D_SZ);
    const float4* wr = reinterpret_cast<const float4*>(Wq    + (size_t)h * D_SZ);
    float acc = 0.f;
    #pragma unroll 4
    for (int i = 0; i < 128; ++i) {
        float4 a = qr[i], w = wr[i];
        acc = fmaf(a.x, w.x, acc); acc = fmaf(a.y, w.y, acc);
        acc = fmaf(a.z, w.z, acc); acc = fmaf(a.w, w.w, acc);
    }
    qp[idx] = acc;
}

// ---------------------------------------------------------------------------
// Kernel 2: scores[b,s] = sum_h w_att[h] * tanh(qp[b,h] + (keys[b,s,:]·Wk[h,:]))
// Split-bf16 WMMA GEMM (hi/lo error compensation) fused with tanh-dot epilogue.
// Workgroup: 128 threads (4 waves), 64 rows. 160 KB LDS -> 2 WG / WGP.
// ---------------------------------------------------------------------------
__global__ __launch_bounds__(128) void scores_kernel(
    const float* __restrict__ keys, const float* __restrict__ Wk,
    const float* __restrict__ qp,   const float* __restrict__ watt,
    float* __restrict__ scores)
{
    __shared__ unsigned short kHi[64 * 512];   // 64 KB
    __shared__ unsigned short kLo[64 * 512];   // 64 KB
    __shared__ unsigned short wHi[16 * 512];   // 16 KB
    __shared__ unsigned short wLo[16 * 512];   // 16 KB

    const int tid     = threadIdx.x;
    const int lane    = tid & 31;
    const int wv      = tid >> 5;              // wave id 0..3
    const int rowBase = blockIdx.x * 64;       // 64-row tiles never cross a batch
    const int b       = rowBase >> 11;         // 2048 rows per batch
    const int n       = lane & 15;
    const int half    = lane >> 4;

    // ---- Stage 64 keys rows (64x512 f32) as bf16 hi/lo planes --------------
    const float4* ksrc = reinterpret_cast<const float4*>(keys + (size_t)rowBase * D_SZ);
    #pragma unroll 4
    for (int i = 0; i < 64; ++i) {
        int idx = tid + i * 128;               // 8192 float4 total
        float4 f = ksrc[idx];
        unsigned short h0 = f2bf_rne(f.x), h1 = f2bf_rne(f.y);
        unsigned short h2 = f2bf_rne(f.z), h3 = f2bf_rne(f.w);
        unsigned short l0 = f2bf_rne(f.x - bf2f(h0)), l1 = f2bf_rne(f.y - bf2f(h1));
        unsigned short l2 = f2bf_rne(f.z - bf2f(h2)), l3 = f2bf_rne(f.w - bf2f(h3));
        *reinterpret_cast<uint2*>(&kHi[idx * 4]) =
            make_uint2((unsigned)h0 | ((unsigned)h1 << 16), (unsigned)h2 | ((unsigned)h3 << 16));
        *reinterpret_cast<uint2*>(&kLo[idx * 4]) =
            make_uint2((unsigned)l0 | ((unsigned)l1 << 16), (unsigned)l2 | ((unsigned)l3 << 16));
    }

    float sAcc[8];
    #pragma unroll
    for (int v = 0; v < 8; ++v) sAcc[v] = 0.f;

    for (int ht = 0; ht < 32; ++ht) {
        __syncthreads();                       // protect Wk tile reuse
        // ---- Stage Wk tile rows [ht*16, ht*16+16) as bf16 hi/lo ------------
        const float4* wsrc = reinterpret_cast<const float4*>(Wk + (size_t)ht * 16 * D_SZ);
        #pragma unroll 4
        for (int i = 0; i < 16; ++i) {
            int idx = tid + i * 128;           // 2048 float4 total
            float4 f = wsrc[idx];
            unsigned short h0 = f2bf_rne(f.x), h1 = f2bf_rne(f.y);
            unsigned short h2 = f2bf_rne(f.z), h3 = f2bf_rne(f.w);
            unsigned short l0 = f2bf_rne(f.x - bf2f(h0)), l1 = f2bf_rne(f.y - bf2f(h1));
            unsigned short l2 = f2bf_rne(f.z - bf2f(h2)), l3 = f2bf_rne(f.w - bf2f(h3));
            *reinterpret_cast<uint2*>(&wHi[idx * 4]) =
                make_uint2((unsigned)h0 | ((unsigned)h1 << 16), (unsigned)h2 | ((unsigned)h3 << 16));
            *reinterpret_cast<uint2*>(&wLo[idx * 4]) =
                make_uint2((unsigned)l0 | ((unsigned)l1 << 16), (unsigned)l2 | ((unsigned)l3 << 16));
        }
        __syncthreads();

        // ---- 16x16 output tile: K=512 in 16 steps of 32, 3 WMMAs each ------
        v8f c = {};
        #pragma unroll 4
        for (int kt = 0; kt < 16; ++kt) {
            const int aRow = (wv * 16 + n) * D_SZ;
            const int aK0  = kt * 32 + half * 8;     // chunks {aK0..+7, aK0+16..+23}
            const int bRow = n * D_SZ;
            const int bK0  = kt * 32 + half * 8;     // chunks {bK0..+7, bK0+16..+23}
            Frag aH, aL, bH, bL;
            aH.u[0] = *reinterpret_cast<const u32x4*>(&kHi[aRow + aK0]);
            aH.u[1] = *reinterpret_cast<const u32x4*>(&kHi[aRow + aK0 + 16]);
            aL.u[0] = *reinterpret_cast<const u32x4*>(&kLo[aRow + aK0]);
            aL.u[1] = *reinterpret_cast<const u32x4*>(&kLo[aRow + aK0 + 16]);
            bH.u[0] = *reinterpret_cast<const u32x4*>(&wHi[bRow + bK0]);
            bH.u[1] = *reinterpret_cast<const u32x4*>(&wHi[bRow + bK0 + 16]);
            bL.u[0] = *reinterpret_cast<const u32x4*>(&wLo[bRow + bK0]);
            bL.u[1] = *reinterpret_cast<const u32x4*>(&wLo[bRow + bK0 + 16]);
            c = __builtin_amdgcn_wmma_f32_16x16x32_bf16(false, aH.v, false, bH.v, (short)0, c, false, false);
            c = __builtin_amdgcn_wmma_f32_16x16x32_bf16(false, aH.v, false, bL.v, (short)0, c, false, false);
            c = __builtin_amdgcn_wmma_f32_16x16x32_bf16(false, aL.v, false, bH.v, (short)0, c, false, false);
        }

        // ---- Fused epilogue: sAcc[m] += w[h] * tanh(q[b,h] + k) (branchless)
        const float qv = qp[(size_t)b * H_SZ + ht * 16 + n];
        const float aw = watt[ht * 16 + n];
        #pragma unroll
        for (int v = 0; v < 8; ++v)
            sAcc[v] += aw * fast_tanh(qv + c[v]);
    }

    // ---- Reduce over N (16 lanes per half-wave), write scores ---------------
    #pragma unroll
    for (int v = 0; v < 8; ++v) {
        float acc = sAcc[v];
        for (int off = 8; off > 0; off >>= 1)
            acc += __shfl_xor(acc, off, 16);
        if (n == 0) {
            int m = v + (half << 3);           // C layout: M = vgpr + 8*(lane/16)
            scores[rowBase + wv * 16 + m] = acc;
        }
    }
}

// ---------------------------------------------------------------------------
// Kernel 3: softmax over S per batch row; writes attn into d_out[16384..]
// ---------------------------------------------------------------------------
__global__ __launch_bounds__(256) void softmax_kernel(
    const float* __restrict__ scores, float* __restrict__ attn)
{
    __shared__ float red[256];
    const int b = blockIdx.x, tid = threadIdx.x;
    const float* row = scores + (size_t)b * S_SZ;
    float v[8];
    float m = -3.402823466e38f;
    #pragma unroll
    for (int i = 0; i < 8; ++i) { v[i] = row[tid + i * 256]; m = fmaxf(m, v[i]); }
    red[tid] = m; __syncthreads();
    for (int s = 128; s > 0; s >>= 1) {
        if (tid < s) red[tid] = fmaxf(red[tid], red[tid + s]);
        __syncthreads();
    }
    m = red[0]; __syncthreads();
    float sum = 0.f;
    #pragma unroll
    for (int i = 0; i < 8; ++i) { v[i] = expf(v[i] - m); sum += v[i]; }
    red[tid] = sum; __syncthreads();
    for (int s = 128; s > 0; s >>= 1) {
        if (tid < s) red[tid] += red[tid + s];
        __syncthreads();
    }
    const float inv = 1.0f / red[0];
    #pragma unroll
    for (int i = 0; i < 8; ++i)
        attn[(size_t)b * S_SZ + tid + i * 256] = v[i] * inv;
}

// ---------------------------------------------------------------------------
// Kernel 4a: partial[b,c,d] = sum_{s in chunk c} attn[b,s] * keys[b,s,d]
// Double-buffered GLOBAL_LOAD_ASYNC_TO_LDS_B128 pipeline (ASYNCcnt-sequenced).
// ---------------------------------------------------------------------------
__global__ __launch_bounds__(256) void wsum_part_kernel(
    const float* __restrict__ keys, const float* __restrict__ attn,
    float* __restrict__ part)
{
    __shared__ float a_s[128];
    const int c = blockIdx.x;                  // 0..15 (chunks of 128 s)
    const int b = blockIdx.y;                  // 0..31
    const int tid = threadIdx.x;
    if (tid < 128) a_s[tid] = attn[(size_t)b * S_SZ + c * 128 + tid];
    const float* kp = keys + ((size_t)b * S_SZ + (size_t)c * 128) * D_SZ;

#if USE_ASYNC
    __shared__ float kbuf[2][16 * D_SZ];       // 2 x 32 KB double buffer
    float acc0 = 0.f, acc1 = 0.f;
    // Preload chunks 0 and 1 (8 async b128 per wave each -> 16 outstanding).
    #pragma unroll
    for (int i = 0; i < 8; ++i) {
        int idx = (tid + i * 256) * 4;
        ASYNC_CP16(&kbuf[0][idx], kp + idx);
    }
    #pragma unroll
    for (int i = 0; i < 8; ++i) {
        int idx = (tid + i * 256) * 4;
        ASYNC_CP16(&kbuf[1][idx], kp + 16 * D_SZ + idx);
    }
    for (int ch = 0; ch < 8; ++ch) {
        // Async ops complete in order: <=8 outstanding => chunk `ch` landed.
        if (ch < 7) { WAIT_ASYNC(8); } else { WAIT_ASYNC(0); }
        __syncthreads();                       // cross-wave visibility
        const float* buf = kbuf[ch & 1];
        #pragma unroll 4
        for (int s = 0; s < 16; ++s) {
            float av = a_s[ch * 16 + s];
            acc0 = fmaf(av, buf[s * D_SZ + tid],       acc0);
            acc1 = fmaf(av, buf[s * D_SZ + tid + 256], acc1);
        }
        __syncthreads();                       // all reads done before overwrite
        if (ch + 2 < 8) {
            const float* src = kp + (size_t)(ch + 2) * 16 * D_SZ;
            #pragma unroll
            for (int i = 0; i < 8; ++i) {
                int idx = (tid + i * 256) * 4;
                ASYNC_CP16(&kbuf[ch & 1][idx], src + idx);
            }
        }
    }
    part[((size_t)b * 16 + c) * D_SZ + tid]       = acc0;
    part[((size_t)b * 16 + c) * D_SZ + tid + 256] = acc1;
#else
    __syncthreads();
    for (int d0 = tid; d0 < D_SZ; d0 += 256) {
        float acc = 0.f;
        #pragma unroll 4
        for (int s = 0; s < 128; ++s)
            acc = fmaf(a_s[s], kp[(size_t)s * D_SZ + d0], acc);
        part[((size_t)b * 16 + c) * D_SZ + d0] = acc;
    }
#endif
}

// ---------------------------------------------------------------------------
// Kernel 4b: out[b,d] = sum_c partial[b,c,d]
// ---------------------------------------------------------------------------
__global__ __launch_bounds__(256) void wsum_final_kernel(
    const float* __restrict__ part, float* __restrict__ out)
{
    const int idx = blockIdx.x * 256 + threadIdx.x;  // 16384
    const int b = idx >> 9, d = idx & 511;
    float acc = 0.f;
    #pragma unroll
    for (int c = 0; c < 16; ++c)
        acc += part[((size_t)b * 16 + c) * D_SZ + d];
    out[idx] = acc;
}

// ---------------------------------------------------------------------------
extern "C" void kernel_launch(void* const* d_in, const int* in_sizes, int n_in,
                              void* d_out, int out_size, void* d_ws, size_t ws_size,
                              hipStream_t stream)
{
    (void)in_sizes; (void)n_in; (void)out_size; (void)ws_size;
    const float* query = (const float*)d_in[0];   // [32, 512]
    const float* keys  = (const float*)d_in[1];   // [32, 2048, 512]
    const float* Wq    = (const float*)d_in[2];   // [512, 512]
    const float* Wk    = (const float*)d_in[3];   // [512, 512]
    const float* watt  = (const float*)d_in[4];   // [512]

    float* out  = (float*)d_out;                  // [32, 512]
    float* attn = out + (size_t)B_SZ * D_SZ;      // [32, 2048] (2nd tuple output)

    float* ws     = (float*)d_ws;
    float* qp     = ws;                           // 32*512      = 16384 floats
    float* scores = qp + (size_t)B_SZ * H_SZ;     // 32*2048     = 65536 floats
    float* part   = scores + (size_t)B_SZ * S_SZ; // 32*16*512   = 262144 floats

    qproj_kernel   <<<64,   256, 0, stream>>>(query, Wq, qp);
    scores_kernel  <<<1024, 128, 0, stream>>>(keys, Wk, qp, watt, scores);
    softmax_kernel <<<32,   256, 0, stream>>>(scores, attn);
    wsum_part_kernel <<<dim3(16, 32), 256, 0, stream>>>(keys, attn, part);
    wsum_final_kernel<<<64,  256, 0, stream>>>(part, out);
}
